// TriangleAttentionStartingNode_36172214566964
// MI455X (gfx1250) — matrix operand order repaired and checked by reading, not compiled, not run on hardware
//
#include <hip/hip_runtime.h>
#include <hip/hip_bf16.h>

// ---------------------------------------------------------------------------
// Triangle attention (starting node) for gfx1250.
// N=256, C_IN=128, H=4, c=32.  All GEMMs via v_wmma_f32_16x16x32_bf16.
// ---------------------------------------------------------------------------

typedef __attribute__((ext_vector_type(16))) __bf16 v16bf;
typedef __attribute__((ext_vector_type(8)))  float  v8f;

union FragU { v16bf v; uint4 u[2]; };

static __device__ __forceinline__ unsigned short f2bf(float f) {
  union { float f; unsigned u; } c; c.f = f;
  unsigned r = c.u + 0x7fffu + ((c.u >> 16) & 1u);   // round-to-nearest-even
  return (unsigned short)(r >> 16);
}
static __device__ __forceinline__ float bf2f(unsigned short h) {
  union { unsigned u; float f; } c; c.u = ((unsigned)h) << 16;
  return c.f;
}

// A operand: row-major src [row][k], 16-bit. Lane L (L<16) holds row L,
// K = {0..7, 16..23}; lane L+16 holds row L, K = {8..15, 24..31}.
static __device__ __forceinline__ v16bf load_frag_a(const unsigned short* p,
                                                    int stride, int kbase) {
  const int lane = threadIdx.x & 31;
  const int hi   = lane >> 4;
  const unsigned short* s = p + (lane & 15) * stride + kbase + hi * 8;
  FragU f;
  f.u[0] = *(const uint4*)(s);
  f.u[1] = *(const uint4*)(s + 16);
  return f.v;
}
// B operand: src is B^T row-major [n][k]. Lane n (<16) holds col n,
// K = {0..7, 16..23}; lane n+16 holds col n, K = {8..15, 24..31}.
static __device__ __forceinline__ v16bf load_frag_b(const unsigned short* p,
                                                    int stride, int kbase) {
  const int lane = threadIdx.x & 31;
  const int hi   = lane >> 4;
  const unsigned short* s = p + (lane & 15) * stride + kbase + hi * 8;
  FragU f;
  f.u[0] = *(const uint4*)(s);
  f.u[1] = *(const uint4*)(s + 16);
  return f.v;
}

#define WMMA_BF16(a, b, c) \
  __builtin_amdgcn_wmma_f32_16x16x32_bf16(false, (a), false, (b), (short)0, (c), false, false)

#define NN   256
#define CIN  128
#define NH   4
#define CHID 32

// ---------------------------------------------------------------------------
// Kernel 0: pack weights -> bf16, transposed to [out_col][k] layout.
// wt: concatenated [wq|wk|wv|wg] -> [512][128]; wot: wo^T -> [128][128].
// ---------------------------------------------------------------------------
__global__ __launch_bounds__(256) void pack_weights_kernel(
    const float* __restrict__ wq, const float* __restrict__ wk,
    const float* __restrict__ wv, const float* __restrict__ wg,
    const float* __restrict__ wo,
    unsigned short* __restrict__ wt, unsigned short* __restrict__ wot) {
  int idx = blockIdx.x * 256 + threadIdx.x;
  if (idx < 512 * 128) {
    int col = idx >> 7, k = idx & 127;
    int grp = col >> 7, cc = col & 127;
    const float* W = (grp == 0) ? wq : (grp == 1) ? wk : (grp == 2) ? wv : wg;
    wt[col * 128 + k] = f2bf(W[k * 128 + cc]);
  } else {
    int r = idx - 512 * 128;
    if (r < 128 * 128) {
      int col = r >> 7, k = r & 127;
      wot[col * 128 + k] = f2bf(wo[k * 128 + col]);
    }
  }
}

// ---------------------------------------------------------------------------
// Kernel 1: LayerNorm + q/k/v/g projections (WMMA) + pair bias nb.
// Block = 256 threads (8 waves), handles 16 grid positions.
//   qb: bf16 [i][h][j][c]   (q * c^-0.5)
//   kb: bf16 [i][h][j][c]
//   vt: bf16 [i][h][c][j]   (transposed for B-operand use)
//   gb: bf16 [pos][128]     sigmoid(x@wg + bg)
//   nbuf: f32 [h][q][k]
// ---------------------------------------------------------------------------
__global__ __launch_bounds__(256) void ln_proj_kernel(
    const float* __restrict__ act, const float* __restrict__ lnw,
    const float* __restrict__ lnb, const float* __restrict__ w2d,
    const float* __restrict__ bg,
    unsigned short* __restrict__ qb, unsigned short* __restrict__ kb,
    unsigned short* __restrict__ vt, unsigned short* __restrict__ gb,
    float* __restrict__ nbuf, const unsigned short* __restrict__ wt) {
  __shared__ unsigned short xl[16][128];
  const int tid = threadIdx.x;
  const int wid = tid >> 5, lane = tid & 31, hi = lane >> 4, nl = lane & 15;
  const int base = blockIdx.x * 16;

  // --- LayerNorm: one wave per 2 rows, lane-parallel over 128 channels ---
  for (int rr = 0; rr < 2; ++rr) {
    const int r = wid * 2 + rr;
    const int pos = base + r;
    const float4 a4 = ((const float4*)(act + (size_t)pos * CIN))[lane];
    float s = a4.x + a4.y + a4.z + a4.w;
    float s2 = a4.x * a4.x + a4.y * a4.y + a4.z * a4.z + a4.w * a4.w;
    for (int m = 16; m >= 1; m >>= 1) {
      s += __shfl_xor(s, m, 32);
      s2 += __shfl_xor(s2, m, 32);
    }
    const float mean = s * (1.0f / 128.0f);
    const float var = s2 * (1.0f / 128.0f) - mean * mean;
    const float rstd = rsqrtf(var + 1e-5f);
    const int ch = lane * 4;
    const float av[4] = {a4.x, a4.y, a4.z, a4.w};
    float p0 = 0.f, p1 = 0.f, p2 = 0.f, p3 = 0.f;
    for (int q = 0; q < 4; ++q) {
      float x = (av[q] - mean) * rstd * lnw[ch + q] + lnb[ch + q];
      xl[r][ch + q] = f2bf(x);
      const float* wrow = w2d + (ch + q) * NH;
      p0 += x * wrow[0]; p1 += x * wrow[1]; p2 += x * wrow[2]; p3 += x * wrow[3];
    }
    for (int m = 16; m >= 1; m >>= 1) {
      p0 += __shfl_xor(p0, m, 32); p1 += __shfl_xor(p1, m, 32);
      p2 += __shfl_xor(p2, m, 32); p3 += __shfl_xor(p3, m, 32);
    }
    if (lane == 0) {
      const int i = pos >> 8, j = pos & 255;
      nbuf[(size_t)0 * 65536 + i * 256 + j] = p0;
      nbuf[(size_t)1 * 65536 + i * 256 + j] = p1;
      nbuf[(size_t)2 * 65536 + i * 256 + j] = p2;
      nbuf[(size_t)3 * 65536 + i * 256 + j] = p3;
    }
  }
  __syncthreads();

  // --- Projections: [16 x 128] @ [128 x 512] via WMMA (8 waves x 4 tiles) ---
  v8f acc[4] = {};
  for (int kk = 0; kk < 4; ++kk) {
    const int kbase = kk * 32;
    v16bf a = load_frag_a(&xl[0][0], 128, kbase);
    for (int t = 0; t < 4; ++t) {
      const int tile = wid * 4 + t;
      v16bf b = load_frag_b(wt + tile * 16 * 128, 128, kbase);
      acc[t] = WMMA_BF16(a, b, acc[t]);
    }
  }
  for (int t = 0; t < 4; ++t) {
    const int tile = wid * 4 + t;
    const int col = tile * 16 + nl;
    const int grp = col >> 7, c128 = col & 127;
    const int h = c128 >> 5, cc = c128 & 31;
    const float bgv = (grp == 3) ? bg[c128] : 0.0f;
    for (int r = 0; r < 8; ++r) {
      const int pos = base + r + hi * 8;
      const int i = pos >> 8, j = pos & 255;
      const float val = acc[t][r];
      if (grp == 0)
        qb[((size_t)(i * NH + h) * NN + j) * CHID + cc] = f2bf(val * 0.17677669529663689f);
      else if (grp == 1)
        kb[((size_t)(i * NH + h) * NN + j) * CHID + cc] = f2bf(val);
      else if (grp == 2)
        vt[((size_t)(i * NH + h) * CHID + cc) * NN + j] = f2bf(val);
      else
        gb[(size_t)pos * CIN + c128] = f2bf(1.0f / (1.0f + __expf(-(val + bgv))));
    }
  }
}

// ---------------------------------------------------------------------------
// Kernel 2: attention per (i, 64-row q tile) + gate + output projection.
// 256 threads (8 waves). Dynamic LDS: Sf 64KB | Pb 32KB | ogf 32KB | ogb 16KB.
// ---------------------------------------------------------------------------
__global__ __launch_bounds__(256) void attn_kernel(
    const float* __restrict__ mask,
    const unsigned short* __restrict__ qb, const unsigned short* __restrict__ kb,
    const unsigned short* __restrict__ vt, const unsigned short* __restrict__ gb,
    const float* __restrict__ nbuf, const unsigned short* __restrict__ wot,
    const float* __restrict__ bo, float* __restrict__ out) {
  extern __shared__ char smem[];
  float*          Sf  = (float*)smem;                               // [64][256]
  unsigned short* Pb  = (unsigned short*)(smem + 65536);            // [64][256]
  float*          ogf = (float*)(smem + 65536 + 32768);             // [64][128]
  unsigned short* ogb = (unsigned short*)(smem + 65536 + 65536);    // [64][128]

  const int tid = threadIdx.x;
  const int wid = tid >> 5, lane = tid & 31, hi = lane >> 4, nl = lane & 15;
  const int i = blockIdx.x >> 2;
  const int q0 = (blockIdx.x & 3) * 64;

  for (int h = 0; h < NH; ++h) {
    const size_t hb = (size_t)(i * NH + h);

    // ---- S = (q * scale) @ k^T : 4 q-tiles x 16 k-tiles, one K-step ----
    {
      const int qt = wid >> 1;
      const unsigned short* qrow = qb + (hb * NN + q0 + qt * 16) * CHID;
      v16bf a = load_frag_a(qrow, CHID, 0);
      for (int t = 0; t < 8; ++t) {
        const int kt = (wid & 1) * 8 + t;
        const unsigned short* krow = kb + (hb * NN + kt * 16) * CHID;
        v16bf b = load_frag_b(krow, CHID, 0);
        v8f acc = {};
        acc = WMMA_BF16(a, b, acc);
        for (int r = 0; r < 8; ++r)
          Sf[(qt * 16 + r + hi * 8) * 256 + kt * 16 + nl] = acc[r];
      }
    }
    __syncthreads();

    // ---- softmax over k with mask bias + pair bias nb (fp32) ----
    for (int rr = 0; rr < 8; ++rr) {
      const int row = wid * 8 + rr;
      const int jq = q0 + row;
      float vb[8];
      float mx = -1e30f;
      for (int u = 0; u < 8; ++u) {
        const int k = u * 32 + lane;
        float vv = Sf[row * 256 + k] + nbuf[(size_t)h * 65536 + jq * 256 + k] +
                   1e9f * (mask[(size_t)i * 256 + k] - 1.0f);
        vb[u] = vv;
        mx = fmaxf(mx, vv);
      }
      for (int m = 16; m >= 1; m >>= 1) mx = fmaxf(mx, __shfl_xor(mx, m, 32));
      float sum = 0.0f;
      for (int u = 0; u < 8; ++u) {
        float e = __expf(vb[u] - mx);
        vb[u] = e;
        sum += e;
      }
      for (int m = 16; m >= 1; m >>= 1) sum += __shfl_xor(sum, m, 32);
      const float inv = 1.0f / sum;
      for (int u = 0; u < 8; ++u)
        Pb[row * 256 + u * 32 + lane] = f2bf(vb[u] * inv);
    }
    __syncthreads();

    // ---- O = P @ V : 4 q-tiles x 2 c-tiles, 8 K-steps ----
    {
      const int qt = wid >> 1;
      const int ct = wid & 1;
      v8f acc = {};
      const unsigned short* prow = Pb + (qt * 16) * 256;
      const unsigned short* vrow = vt + (hb * CHID + ct * 16) * NN;
      for (int ks = 0; ks < 8; ++ks) {
        v16bf a = load_frag_a(prow, 256, ks * 32);
        v16bf b = load_frag_b(vrow, 256, ks * 32);
        acc = WMMA_BF16(a, b, acc);
      }
      for (int r = 0; r < 8; ++r)
        ogf[(qt * 16 + r + hi * 8) * CIN + h * CHID + ct * 16 + nl] = acc[r];
    }
    __syncthreads();
  }

  // ---- gate: og = O * sigmoid(x@wg + bg), convert to bf16 ----
  for (int idx = tid; idx < 64 * CIN; idx += 256) {
    const int row = idx >> 7, col = idx & 127;
    const size_t pos = (size_t)i * NN + q0 + row;
    ogb[idx] = f2bf(ogf[idx] * bf2f(gb[pos * CIN + col]));
  }
  __syncthreads();

  // ---- out = og @ wo + bo : 4 q-tiles x 8 n-tiles, 4 K-steps ----
  {
    const int qt = wid >> 1;
    v8f acc[4] = {};
    for (int kk = 0; kk < 4; ++kk) {
      v16bf a = load_frag_a(ogb + qt * 16 * CIN, CIN, kk * 32);
      for (int t = 0; t < 4; ++t) {
        const int nt = (wid & 1) * 4 + t;
        v16bf b = load_frag_b(wot + nt * 16 * 128, 128, kk * 32);
        acc[t] = WMMA_BF16(a, b, acc[t]);
      }
    }
    for (int t = 0; t < 4; ++t) {
      const int nt = (wid & 1) * 4 + t;
      const int col = nt * 16 + nl;
      const float bov = bo[col];
      for (int r = 0; r < 8; ++r) {
        const int row = qt * 16 + r + hi * 8;
        const size_t pos = (size_t)i * NN + q0 + row;
        out[pos * CIN + col] = acc[t][r] + bov;
      }
    }
  }
}

// ---------------------------------------------------------------------------
extern "C" void kernel_launch(void* const* d_in, const int* in_sizes, int n_in,
                              void* d_out, int out_size, void* d_ws, size_t ws_size,
                              hipStream_t stream) {
  const float* act  = (const float*)d_in[0];
  const float* mask = (const float*)d_in[1];
  const float* lnw  = (const float*)d_in[2];
  const float* lnb  = (const float*)d_in[3];
  const float* w2d  = (const float*)d_in[4];
  const float* wq   = (const float*)d_in[5];
  const float* wk   = (const float*)d_in[6];
  const float* wv   = (const float*)d_in[7];
  const float* wg   = (const float*)d_in[8];
  const float* bg   = (const float*)d_in[9];
  const float* wo   = (const float*)d_in[10];
  const float* bo   = (const float*)d_in[11];
  float* out = (float*)d_out;

  char* ws = (char*)d_ws;
  unsigned short* qb   = (unsigned short*)(ws);                          // 16 MB
  unsigned short* kb   = (unsigned short*)(ws + (size_t)16 * 1048576);   // 16 MB
  unsigned short* vt   = (unsigned short*)(ws + (size_t)32 * 1048576);   // 16 MB
  unsigned short* gb   = (unsigned short*)(ws + (size_t)48 * 1048576);   // 16 MB
  float*          nbuf = (float*)         (ws + (size_t)64 * 1048576);   // 1 MB
  unsigned short* wt   = (unsigned short*)(ws + (size_t)65 * 1048576);   // 128 KB
  unsigned short* wot  = (unsigned short*)(ws + (size_t)65 * 1048576 + 131072); // 32 KB

  (void)in_sizes; (void)n_in; (void)out_size; (void)ws_size;

  const int ATTN_LDS = 65536 + 32768 + 32768 + 16384;  // 144 KB
  hipFuncSetAttribute((const void*)attn_kernel,
                      hipFuncAttributeMaxDynamicSharedMemorySize, ATTN_LDS);

  pack_weights_kernel<<<(512 * 128 + 128 * 128) / 256, 256, 0, stream>>>(
      wq, wk, wv, wg, wo, wt, wot);
  ln_proj_kernel<<<(256 * 256) / 16, 256, 0, stream>>>(
      act, lnw, lnb, w2d, bg, qb, kb, vt, gb, nbuf, wt);
  attn_kernel<<<256 * 4, 256, ATTN_LDS, stream>>>(
      mask, qb, kb, vt, gb, nbuf, wot, bo, out);
}
